// AttentionBlock_28303834481261
// MI455X (gfx1250) — matrix operand order repaired and checked by reading, not compiled, not run on hardware
//
#include <hip/hip_runtime.h>

typedef _Float16 half_t;
typedef __attribute__((ext_vector_type(16))) _Float16 v16h;
typedef __attribute__((ext_vector_type(8)))  _Float16 v8h;
typedef __attribute__((ext_vector_type(8)))  float    v8f;

#define BATCH 16
#define CCH   512
#define NH    8
#define DH    64
#define HWSZ  1024

// ---------------------------------------------------------------------------
// WMMA helper: D = A(16x32 f16) * B(32x16 f16) + C(16x16 f32)
// ---------------------------------------------------------------------------
__device__ __forceinline__ v8f wmma16(v16h a, v16h b, v8f c) {
  return __builtin_amdgcn_wmma_f32_16x16x32_f16(
      false, a, false, b, (short)0, c, false, false);
}

// A fragment (16x32, row-major source, K contiguous, row stride lda halves).
// ISA 7.12.2: lanes 0-15 -> M=lane, VGPR0-3: K = half*8 + 0..7, VGPR4-7: +16.
__device__ __forceinline__ v16h frag_a(const half_t* p, int lda, int lane) {
  const int m = lane & 15, h = lane >> 4;
  const half_t* r = p + (size_t)m * lda + h * 8;
  v8h lo = *(const v8h*)r;
  v8h hi = *(const v8h*)(r + 16);
  v16h out;
#pragma unroll
  for (int i = 0; i < 8; ++i) { out[i] = lo[i]; out[i + 8] = hi[i]; }
  return out;
}

// B fragment (32x16): source stored N-major (row = N, K contiguous, stride ldb).
// Elements 0-7: K = half*8 + 0..7, elements 8-15: +16; lane n = lane&15 holds column n.
__device__ __forceinline__ v16h frag_b(const half_t* p, int ldb, int lane) {
  const int n = lane & 15, h = lane >> 4;
  const half_t* r = p + (size_t)n * ldb + h * 8;
  v8h lo = *(const v8h*)r;
  v8h hi = *(const v8h*)(r + 16);
  v16h out;
#pragma unroll
  for (int i = 0; i < 8; ++i) { out[i] = lo[i]; out[i + 8] = hi[i]; }
  return out;
}

// Async global -> LDS 16B copy (per lane) + wait, per CDNA5 ISA
__device__ __forceinline__ void async_copy_b128(unsigned lds_addr, const void* gaddr) {
  asm volatile("global_load_async_to_lds_b128 %0, %1, off"
               :: "v"(lds_addr), "v"(gaddr) : "memory");
}
__device__ __forceinline__ void wait_async0() {
  asm volatile("s_wait_asynccnt 0" ::: "memory");
}

// ---------------------------------------------------------------------------
// 1) GroupNorm: x[b][c][hw] fp32 -> xnT[b][hw][c] f16 (transposed for GEMM-B)
// ---------------------------------------------------------------------------
__global__ void gn_kernel(const float* __restrict__ x, const float* __restrict__ w,
                          const float* __restrict__ bgn, half_t* __restrict__ xnT) {
  const int b = blockIdx.x >> 3, g = blockIdx.x & 7;
  const float* xp = x + ((size_t)b * CCH + g * 64) * HWSZ;
  float s = 0.f, ss = 0.f;
  for (int e = threadIdx.x; e < 64 * HWSZ; e += 256) {
    float v = xp[e]; s += v; ss += v * v;
  }
  __shared__ float red[512];
  red[threadIdx.x] = s; red[256 + threadIdx.x] = ss;
  __syncthreads();
  for (int off = 128; off > 0; off >>= 1) {
    if ((int)threadIdx.x < off) {
      red[threadIdx.x]       += red[threadIdx.x + off];
      red[256 + threadIdx.x] += red[256 + threadIdx.x + off];
    }
    __syncthreads();
  }
  const float mean = red[0] * (1.f / 65536.f);
  const float var  = red[256] * (1.f / 65536.f) - mean * mean;
  const float inv  = rsqrtf(var + 1e-5f);
  for (int e = threadIdx.x; e < 64 * HWSZ; e += 256) {
    const int cl = e >> 10, hw = e & 1023, c = g * 64 + cl;
    const float v = (xp[e] - mean) * inv * w[c] + bgn[c];
    xnT[((size_t)b * HWSZ + hw) * CCH + c] = (half_t)v;
  }
}

// ---------------------------------------------------------------------------
// 2) Weight conversion fp32 -> f16
// ---------------------------------------------------------------------------
__global__ void cvt_w_kernel(const float* __restrict__ qw, const float* __restrict__ pw,
                             half_t* __restrict__ q16, half_t* __restrict__ p16) {
  const int nq = 3 * CCH * CCH, np = CCH * CCH;
  for (int i = blockIdx.x * blockDim.x + threadIdx.x; i < nq + np;
       i += gridDim.x * blockDim.x) {
    if (i < nq) q16[i] = (half_t)qw[i];
    else        p16[i - nq] = (half_t)pw[i - nq];
  }
}

// ---------------------------------------------------------------------------
// 3) QKV GEMM, 2x2 register tiling (32x32 output per wave).
//    q,k scattered transposed: qT/kT[b][h][hw][d]; v natural: v[b][h][d][hw]
// ---------------------------------------------------------------------------
__global__ void qkv_gemm_kernel(const half_t* __restrict__ w16, const float* __restrict__ bias,
                                const half_t* __restrict__ xnT,
                                half_t* __restrict__ qT, half_t* __restrict__ kT,
                                half_t* __restrict__ vbuf) {
  const int lane = threadIdx.x & 31;
  const int wid  = blockIdx.x * 8 + (threadIdx.x >> 5);
  const int b  = wid / (48 * 32);          // M tiles: 1536/32=48, N tiles: 1024/32=32
  const int r0 = wid % (48 * 32);
  const int mt = r0 / 32, nt = r0 % 32;
  const int o0 = mt * 32, hw0 = nt * 32;
  const int hf = lane >> 4;

  v8f acc[2][2];
#pragma unroll
  for (int mi = 0; mi < 2; ++mi)
#pragma unroll
    for (int ni = 0; ni < 2; ++ni)
#pragma unroll
      for (int r = 0; r < 8; ++r)
        acc[mi][ni][r] = bias[o0 + mi * 16 + 8 * hf + r];

  const half_t* wbase = w16 + (size_t)o0 * CCH;
  const half_t* xbase = xnT + ((size_t)b * HWSZ + hw0) * CCH;
#pragma unroll 2
  for (int k0 = 0; k0 < CCH; k0 += 32) {
    v16h a0 = frag_a(wbase + k0, CCH, lane);
    v16h a1 = frag_a(wbase + (size_t)16 * CCH + k0, CCH, lane);
    v16h b0 = frag_b(xbase + k0, CCH, lane);
    v16h b1 = frag_b(xbase + (size_t)16 * CCH + k0, CCH, lane);
    acc[0][0] = wmma16(a0, b0, acc[0][0]);
    acc[0][1] = wmma16(a0, b1, acc[0][1]);
    acc[1][0] = wmma16(a1, b0, acc[1][0]);
    acc[1][1] = wmma16(a1, b1, acc[1][1]);
  }

#pragma unroll
  for (int mi = 0; mi < 2; ++mi) {
    const int om = o0 + mi * 16;
    const int which = om / CCH;
    const int h   = (om % CCH) / DH;
    const int dd0 = om % DH;
#pragma unroll
    for (int ni = 0; ni < 2; ++ni) {
      const int hwn = hw0 + ni * 16;
      if (which < 2) {
        half_t* dst = (which == 0) ? qT : kT;  // [b][h][hw][d]: contiguous 8 halves
        const size_t base =
            (((size_t)b * NH + h) * HWSZ + hwn + (lane & 15)) * DH + dd0 + 8 * hf;
        v8h outv;
#pragma unroll
        for (int r = 0; r < 8; ++r) outv[r] = (half_t)acc[mi][ni][r];
        *(v8h*)(dst + base) = outv;
      } else {                                 // v[b][h][d][hw]: strided scatter
        const size_t base =
            (((size_t)b * NH + h) * DH + dd0) * HWSZ + hwn + (lane & 15);
#pragma unroll
        for (int r = 0; r < 8; ++r)
          vbuf[base + (size_t)(r + 8 * hf) * HWSZ] = (half_t)acc[mi][ni][r];
      }
    }
  }
}

// ---------------------------------------------------------------------------
// 4) Flash attention with block-shared K/V tiles staged to LDS via async copy.
//    Block = one (b,h), 128 i-rows (8 waves x 16). j streamed in steps of 32.
// ---------------------------------------------------------------------------
__global__ void attn_kernel(const half_t* __restrict__ qT, const half_t* __restrict__ kT,
                            const half_t* __restrict__ vbuf, half_t* __restrict__ oT) {
  const int tid  = threadIdx.x;
  const int lane = tid & 31;
  const int wv   = tid >> 5;
  const int iblk = blockIdx.x & 7;
  const int bh   = blockIdx.x >> 3;
  const int b = bh >> 3, h = bh & 7;
  const int i0 = iblk * 128 + wv * 16;
  const int hf = lane >> 4;

  const half_t* qb = qT   + (size_t)bh * HWSZ * DH;
  const half_t* kb = kT   + (size_t)bh * HWSZ * DH;
  const half_t* vb = vbuf + (size_t)bh * DH * HWSZ;

  __shared__ __align__(16) half_t ktile[32 * 64];   // rows j, cols d   (4 KB)
  __shared__ __align__(16) half_t vtile[64 * 32];   // rows d, cols j   (4 KB)
  __shared__ __align__(16) half_t pstage[8 * 16 * 32];
  half_t* pw = pstage + wv * 512;

  // Per-thread 16B slices of the shared K/V tiles (256 threads cover both).
  const int krow = tid >> 3, kseg = tid & 7;                 // 32 x (8*16B)
  const half_t* kgsrc = kb + (size_t)krow * DH + kseg * 8;
  const unsigned klds = (unsigned)(uintptr_t)(ktile + krow * 64 + kseg * 8);
  const int vrow = tid >> 2, vseg = tid & 3;                 // 64 x (4*16B)
  const half_t* vgsrc = vb + (size_t)vrow * HWSZ + vseg * 8;
  const unsigned vlds = (unsigned)(uintptr_t)(vtile + vrow * 32 + vseg * 8);

  const v16h aq0 = frag_a(qb + (size_t)i0 * DH, DH, lane);       // K = d 0..31
  const v16h aq1 = frag_a(qb + (size_t)i0 * DH + 32, DH, lane);  // K = d 32..63

  float mrow[8], lrow[8];
#pragma unroll
  for (int r = 0; r < 8; ++r) { mrow[r] = -3.0e38f; lrow[r] = 0.f; }
  v8f oacc[4];
#pragma unroll
  for (int t = 0; t < 4; ++t) oacc[t] = (v8f){};

  for (int j0 = 0; j0 < HWSZ; j0 += 32) {
    // Stage K (rows j0..j0+31) and V (cols j0..j0+31) tiles into LDS.
    async_copy_b128(klds, kgsrc + (size_t)j0 * DH);
    async_copy_b128(vlds, vgsrc + j0);
    wait_async0();
    __syncthreads();

    v16h bk0  = frag_b(ktile, 64, lane);
    v16h bk0h = frag_b(ktile + 32, 64, lane);
    v16h bk1  = frag_b(ktile + 16 * 64, 64, lane);
    v16h bk1h = frag_b(ktile + 16 * 64 + 32, 64, lane);
    v8f s0 = (v8f){}, s1 = (v8f){};
    s0 = wmma16(aq0, bk0, s0); s0 = wmma16(aq1, bk0h, s0);
    s1 = wmma16(aq0, bk1, s1); s1 = wmma16(aq1, bk1h, s1);

#pragma unroll
    for (int r = 0; r < 8; ++r) {
      const float a = s0[r] * 0.125f;            // scale = d^-0.5 = 1/8
      const float c = s1[r] * 0.125f;
      float mx = fmaxf(a, c);
#pragma unroll
      for (int m = 1; m <= 8; m <<= 1) mx = fmaxf(mx, __shfl_xor(mx, m, 32));
      const float mnew = fmaxf(mrow[r], mx);
      const float corr = __expf(mrow[r] - mnew);
      const float p0 = __expf(a - mnew);
      const float p1 = __expf(c - mnew);
      float rs = p0 + p1;
#pragma unroll
      for (int m = 1; m <= 8; m <<= 1) rs += __shfl_xor(rs, m, 32);
      lrow[r] = lrow[r] * corr + rs;
      mrow[r] = mnew;
#pragma unroll
      for (int t = 0; t < 4; ++t) oacc[t][r] *= corr;
      const int ml = r + 8 * hf;
      pw[ml * 32 + (lane & 15)]      = (half_t)p0;
      pw[ml * 32 + 16 + (lane & 15)] = (half_t)p1;
    }
    // Reshape P: D-fragment -> A-fragment through wave-private LDS.
    const v16h ap = frag_a(pw, 32, lane);
#pragma unroll
    for (int t = 0; t < 4; ++t) {
      v16h bv = frag_b(vtile + (size_t)(t * 16) * 32, 32, lane);
      oacc[t] = wmma16(ap, bv, oacc[t]);
    }
    __syncthreads();   // all waves done with ktile/vtile before next stage
  }

#pragma unroll
  for (int t = 0; t < 4; ++t) {
#pragma unroll
    for (int r = 0; r < 8; ++r) {
      const float v = oacc[t][r] / lrow[r];
      const int i = i0 + r + 8 * hf;
      const int c = h * DH + t * 16 + (lane & 15);
      oT[((size_t)b * HWSZ + i) * CCH + c] = (half_t)v;
    }
  }
}

// ---------------------------------------------------------------------------
// 5) Proj GEMM + residual, 2x2 register tiling:
//    out[b][co][hw] = oT[b][hw][co] + b[co] + sum_c pw[co][c]*oT[b][hw][c]
// ---------------------------------------------------------------------------
__global__ void proj_gemm_kernel(const half_t* __restrict__ p16, const float* __restrict__ bias,
                                 const half_t* __restrict__ oT, float* __restrict__ out) {
  const int lane = threadIdx.x & 31;
  const int wid  = blockIdx.x * 8 + (threadIdx.x >> 5);
  const int b  = wid / (16 * 32);          // M tiles: 512/32=16, N tiles: 32
  const int r0 = wid % (16 * 32);
  const int mt = r0 / 32, nt = r0 % 32;
  const int co0 = mt * 32, hw0 = nt * 32;
  const int hf = lane >> 4;

  v8f acc[2][2];
#pragma unroll
  for (int mi = 0; mi < 2; ++mi)
#pragma unroll
    for (int ni = 0; ni < 2; ++ni)
#pragma unroll
      for (int r = 0; r < 8; ++r)
        acc[mi][ni][r] = bias[co0 + mi * 16 + 8 * hf + r];

  const half_t* wbase = p16 + (size_t)co0 * CCH;
  const half_t* obase = oT + ((size_t)b * HWSZ + hw0) * CCH;
#pragma unroll 2
  for (int k0 = 0; k0 < CCH; k0 += 32) {
    v16h a0 = frag_a(wbase + k0, CCH, lane);
    v16h a1 = frag_a(wbase + (size_t)16 * CCH + k0, CCH, lane);
    v16h b0 = frag_b(obase + k0, CCH, lane);
    v16h b1 = frag_b(obase + (size_t)16 * CCH + k0, CCH, lane);
    acc[0][0] = wmma16(a0, b0, acc[0][0]);
    acc[0][1] = wmma16(a0, b1, acc[0][1]);
    acc[1][0] = wmma16(a1, b0, acc[1][0]);
    acc[1][1] = wmma16(a1, b1, acc[1][1]);
  }

#pragma unroll
  for (int mi = 0; mi < 2; ++mi) {
#pragma unroll
    for (int ni = 0; ni < 2; ++ni) {
      const int com = co0 + mi * 16, hwn = hw0 + ni * 16;
      const v8h res = *(const v8h*)(oT + ((size_t)b * HWSZ + hwn + (lane & 15)) * CCH
                                        + com + 8 * hf);
      const size_t ob = ((size_t)b * CCH + com + 8 * hf) * HWSZ + hwn + (lane & 15);
#pragma unroll
      for (int r = 0; r < 8; ++r)
        out[ob + (size_t)r * HWSZ] = acc[mi][ni][r] + (float)res[r];
    }
  }
}

// ---------------------------------------------------------------------------
extern "C" void kernel_launch(void* const* d_in, const int* in_sizes, int n_in,
                              void* d_out, int out_size, void* d_ws, size_t ws_size,
                              hipStream_t stream) {
  (void)in_sizes; (void)n_in; (void)out_size; (void)ws_size;
  const float* x      = (const float*)d_in[0];
  const float* gn_w   = (const float*)d_in[1];
  const float* gn_b   = (const float*)d_in[2];
  const float* qkv_w  = (const float*)d_in[3];
  const float* qkv_b  = (const float*)d_in[4];
  const float* proj_w = (const float*)d_in[5];
  const float* proj_b = (const float*)d_in[6];
  float* out = (float*)d_out;

  half_t* ws   = (half_t*)d_ws;
  half_t* w16q = ws;                   // 3*512*512      = 786432 halves
  half_t* w16p = w16q + 786432;        // 512*512        = 262144
  half_t* xnT  = w16p + 262144;        // 16*1024*512    = 8388608
  half_t* qT   = xnT  + 8388608;       // [b][h][hw][d]
  half_t* kT   = qT   + 8388608;
  half_t* vbuf = kT   + 8388608;       // [b][h][d][hw]
  half_t* oT   = vbuf + 8388608;       // [b][hw][c]     (total ~82 MB)

  cvt_w_kernel   <<<512,  256, 0, stream>>>(qkv_w, proj_w, w16q, w16p);
  gn_kernel      <<<128,  256, 0, stream>>>(x, gn_w, gn_b, xnT);
  qkv_gemm_kernel<<<3072, 256, 0, stream>>>(w16q, qkv_b, xnT, qT, kT, vbuf);
  attn_kernel    <<<1024, 256, 0, stream>>>(qT, kT, vbuf, oT);
  proj_gemm_kernel<<<1024, 256, 0, stream>>>(w16p, proj_b, oT, out);
}
